// _InfoNCETemporalLoss_76836964926183
// MI455X (gfx1250) — matrix operand order repaired and checked by reading, not compiled, not run on hardware
//
#include <hip/hip_runtime.h>
#include <stdint.h>

typedef _Float16 v16h __attribute__((ext_vector_type(16)));
typedef _Float16 v8h  __attribute__((ext_vector_type(8)));
typedef float    v8f  __attribute__((ext_vector_type(8)));

#define B_   8
#define T_   16
#define P_   1024
#define D_   128
#define TEMP_INV 10.0f   // 1 / TEMPERATURE

// ---- CDNA5 async global->LDS DMA (ASYNCcnt-tracked), per ISA 08_async_tensor
__device__ __forceinline__ void async_ld16(uint32_t lds_byte_off, const void* gaddr) {
  // VDST = VGPR holding LDS byte address, VADDR = 64-bit global address
  asm volatile("global_load_async_to_lds_b128 %0, %1, off"
               :: "v"(lds_byte_off), "v"(gaddr)
               : "memory");
}
__device__ __forceinline__ void wait_async0() {
  asm volatile("s_wait_asynccnt 0x0" ::: "memory");
}

// ---------------------------------------------------------------------------
// Kernel 1: L2-normalize each 128-dim point, transpose (B,D,T*P)->(B*T*P, D),
// store as f16 row-major [point][d] in workspace.
// ---------------------------------------------------------------------------
__global__ __launch_bounds__(256) void k_normalize(const float* __restrict__ enc,
                                                   _Float16* __restrict__ ptsn) {
  __shared__ float tile[128][65];   // ~33 KB, stride-65 -> conflict-free columns
  __shared__ float scl[64];
  const int tid    = threadIdx.x;
  const int blk    = blockIdx.x;        // B*T*(P/64) = 2048
  const int pchunk = blk & 15;
  const int bt     = blk >> 4;          // b*T + t
  const int b      = bt >> 4;
  const int t      = bt & 15;

  const long base = (long)b * D_ * (long)(T_ * P_) + (long)t * P_ + (long)pchunk * 64;

  for (int i = tid; i < 128 * 64; i += 256) {
    const int d = i >> 6, p = i & 63;
    tile[d][p] = enc[base + (long)d * (T_ * P_) + p];
  }
  __syncthreads();

  if (tid < 64) {
    float s = 0.0f;
#pragma unroll
    for (int d = 0; d < 128; ++d) { const float v = tile[d][tid]; s += v * v; }
    scl[tid] = 1.0f / fmaxf(sqrtf(s), 1e-12f);
  }
  __syncthreads();

  const long obase = ((long)bt * P_ + (long)pchunk * 64) * D_;
  for (int i = tid; i < 64 * 16; i += 256) {
    const int p = i >> 4, c = i & 15;
    const float s = scl[p];
    v8h h;
#pragma unroll
    for (int j = 0; j < 8; ++j) h[j] = (_Float16)(tile[c * 8 + j][p] * s);
    *(v8h*)(ptsn + obase + (long)p * D_ + c * 8) = h;
  }
}

// ---------------------------------------------------------------------------
// Kernel 2: per (pair, 128-row stripe): sim = An·Bn^T * 10 via
// v_wmma_f32_16x16x32_f16 with async double-buffered B staging in LDS.
// Block = 256 thr = 8 waves; wave w owns rows m_base = mblock*128 + w*16.
// Stage = 32 columns (two 16x16 N-tiles), 32 stages, 1 barrier per stage.
// Per sub-tile: batch all 8 ds_load_b128 (4 B fragments) then 4 back-to-back
// WMMAs so DS latency is paid once, not per WMMA.
// ---------------------------------------------------------------------------
__global__ __launch_bounds__(256) void k_simlse(const _Float16* __restrict__ ptsn,
                                                const uint8_t* __restrict__ mask,
                                                float* __restrict__ partial) {
  __shared__ __align__(16) _Float16 Btile[2][32 * D_];   // 2 x 8 KB
  __shared__ float blocksum;

  const int  tid    = threadIdx.x;
  const int  wave   = tid >> 5;
  const int  lane   = tid & 31;
  const bool hi     = lane >= 16;
  const int  blk    = blockIdx.x;        // pair*8 + mblock, 960 blocks
  const int  mblock = blk & 7;
  const int  pair   = blk >> 3;
  const int  b      = pair / 15;
  const int  t      = pair % 15;

  const _Float16* anchors = ptsn + ((long)(b * T_ + t)     * P_) * D_;
  const _Float16* targets = ptsn + ((long)(b * T_ + t + 1) * P_) * D_;
  const int m_base = mblock * 128 + wave * 16;
  const int mtile  = m_base >> 4;

  // --- A fragments (16 rows x K=128) loaded once, reused for all N-tiles.
  // 16-bit A layout: lanes 0-15 row=lane hold K {0..7,16..23}; lanes 16-31
  // row=lane-16 hold K {8..15,24..31} (per 32-K fragment).
  v16h Af[4];
  {
    const int row = m_base + (lane & 15);
    const _Float16* rp0 = anchors + (long)row * D_ + (hi ? 8 : 0);
#pragma unroll
    for (int kk = 0; kk < 4; ++kk) {
      const _Float16* rp = rp0 + kk * 32;
      const v8h lo8 = *(const v8h*)(rp);
      const v8h hi8 = *(const v8h*)(rp + 16);
      v16h a;
#pragma unroll
      for (int i = 0; i < 8; ++i) { a[i] = lo8[i]; a[8 + i] = hi8[i]; }
      Af[kk] = a;
    }
  }

  // LDS byte offsets of the two staging buffers
  const uint32_t lds0 = (uint32_t)(uintptr_t)(&Btile[0][0]);
  const uint32_t lds1 = (uint32_t)(uintptr_t)(&Btile[1][0]);

  v8f   sumexp = {0, 0, 0, 0, 0, 0, 0, 0};
  float dl     = 0.0f;   // this lane's diagonal contribution (if any)

  // prefetch stage 0 into buffer 0: 8 KB = 512 x 16B chunks, 2 per thread
  {
    const _Float16* src = targets;                 // stage 0, cols 0..31
    async_ld16(lds0 + (uint32_t)tid * 16u,          src + (size_t)tid * 8);
    async_ld16(lds0 + 4096u + (uint32_t)tid * 16u,  src + 2048 + (size_t)tid * 8);
  }

  for (int st = 0; st < 32; ++st) {
    wait_async0();        // my DMA into buf[st&1] complete
    __syncthreads();      // everyone's DMA visible; prev buffer fully consumed

    if (st + 1 < 32) {    // prefetch next stage into the other buffer
      const uint32_t dst = (st & 1) ? lds0 : lds1;
      const _Float16* src = targets + (size_t)(st + 1) * 32 * D_;
      async_ld16(dst + (uint32_t)tid * 16u,          src + (size_t)tid * 8);
      async_ld16(dst + 4096u + (uint32_t)tid * 16u,  src + 2048 + (size_t)tid * 8);
    }

    const _Float16* Bbuf = &Btile[st & 1][0];
#pragma unroll
    for (int sub = 0; sub < 2; ++sub) {
      const int nt = st * 2 + sub;

      // batch-load all 4 B fragments (8 x ds_load_b128, one dscnt wait)
      v16h bfr[4];
#pragma unroll
      for (int kk = 0; kk < 4; ++kk) {
        // B layout: lane-lo col holds K {0..7,16..23} of fragment; lane-hi K {8..15,24..31}
        const _Float16* bp = Bbuf + ((lane & 15) + sub * 16) * D_
                                  + kk * 32 + (hi ? 8 : 0);
        const v8h b0 = *(const v8h*)(bp);
        const v8h b1 = *(const v8h*)(bp + 16);
        v16h bf;
#pragma unroll
        for (int i = 0; i < 8; ++i) { bf[i] = b0[i]; bf[8 + i] = b1[i]; }
        bfr[kk] = bf;
      }

      // 4 back-to-back WMMAs (pipelined D->C accumulation, no mem stalls)
      v8f c = {0, 0, 0, 0, 0, 0, 0, 0};
#pragma unroll
      for (int kk = 0; kk < 4; ++kk)
        c = __builtin_amdgcn_wmma_f32_16x16x32_f16(false, Af[kk], false, bfr[kk],
                                                   (short)0, c, false, false);

      // sim bounded in [-10,10] (unit vectors / 0.1) -> no max-shift needed
#pragma unroll
      for (int r = 0; r < 8; ++r) sumexp[r] += __expf(c[r] * TEMP_INV);

      if (nt == mtile) {   // wave-uniform; diagonal lives in this tile
#pragma unroll
        for (int r = 0; r < 8; ++r) {
          if (lane == r)      dl = c[r] * TEMP_INV;  // row m_base+r   @ lane r
          if (lane == 24 + r) dl = c[r] * TEMP_INV;  // row m_base+8+r @ lane 24+r
        }
      }
    }
  }

  // row-wise reduce over the 16 lanes sharing each row
#pragma unroll
  for (int r = 0; r < 8; ++r) {
    float s = sumexp[r];
    s += __shfl_xor(s, 1);
    s += __shfl_xor(s, 2);
    s += __shfl_xor(s, 4);
    s += __shfl_xor(s, 8);
    sumexp[r] = s;
  }

  // masked ce accumulation: lane 0 -> rows m_base..+7, lane 16 -> +8..+15
  float local = 0.0f;
  const uint8_t* mrow = mask + (long)(b * T_ + t) * P_;
#pragma unroll
  for (int r = 0; r < 8; ++r) {
    const float dlo = __shfl(dl, r);        // diag of row m_base+r
    const float dhi = __shfl(dl, 24 + r);   // diag of row m_base+8+r
    if (lane == 0) {
      if (mrow[m_base + r])     local += __logf(sumexp[r]) - dlo;
    }
    if (lane == 16) {
      if (mrow[m_base + 8 + r]) local += __logf(sumexp[r]) - dhi;
    }
  }

  __syncthreads();
  if (tid == 0) blocksum = 0.0f;
  __syncthreads();
  if (lane == 0 || lane == 16) atomicAdd(&blocksum, local);
  __syncthreads();
  if (tid == 0) partial[blk] = blocksum;    // unique slot: no global atomics
}

// ---------------------------------------------------------------------------
// Kernel 3: combine 8 partials per pair, gate on n_valid>=2, emit scalar.
// ---------------------------------------------------------------------------
__global__ __launch_bounds__(128) void k_finalize(const uint8_t* __restrict__ mask,
                                                  const float* __restrict__ partial,
                                                  float* __restrict__ out) {
  __shared__ float s_tot[128];
  __shared__ float s_cnt[128];
  const int tid = threadIdx.x;
  float tot = 0.0f, cnt = 0.0f;
  if (tid < B_ * (T_ - 1)) {
    const int b = tid / (T_ - 1), t = tid % (T_ - 1);
    const uint8_t* mrow = mask + (long)(b * T_ + t) * P_;
    int nv = 0;
    for (int i = 0; i < P_; ++i) nv += mrow[i] ? 1 : 0;
    float ps = 0.0f;
#pragma unroll
    for (int mb = 0; mb < 8; ++mb) ps += partial[tid * 8 + mb];
    const float pl = ps / fmaxf((float)nv, 1.0f);
    if (nv >= 2) { tot = pl; cnt = 1.0f; }
  }
  s_tot[tid] = tot; s_cnt[tid] = cnt;
  __syncthreads();
  for (int off = 64; off > 0; off >>= 1) {
    if (tid < off) { s_tot[tid] += s_tot[tid + off]; s_cnt[tid] += s_cnt[tid + off]; }
    __syncthreads();
  }
  if (tid == 0) out[0] = s_tot[0] / fmaxf(s_cnt[0], 1.0f);
}

// ---------------------------------------------------------------------------
extern "C" void kernel_launch(void* const* d_in, const int* in_sizes, int n_in,
                              void* d_out, int out_size, void* d_ws, size_t ws_size,
                              hipStream_t stream) {
  const float*   enc  = (const float*)d_in[0];
  const uint8_t* mask = (const uint8_t*)d_in[1];   // jnp bool -> 1 byte/elem
  (void)in_sizes; (void)n_in; (void)out_size; (void)ws_size;

  _Float16* ptsn    = (_Float16*)d_ws;                               // 32 MB
  float*    partial = (float*)((char*)d_ws +
                      (size_t)B_ * T_ * P_ * D_ * sizeof(_Float16)); // 3840 B
  float*    out     = (float*)d_out;

  k_normalize<<<B_ * T_ * (P_ / 64), 256, 0, stream>>>(enc, ptsn);
  k_simlse  <<<B_ * (T_ - 1) * 8,    256, 0, stream>>>(ptsn, mask, partial);
  k_finalize<<<1, 128, 0, stream>>>(mask, partial, out);
}
